// DeepSeekMoE_45268955300112
// MI455X (gfx1250) — matrix operand order repaired and checked by reading, not compiled, not run on hardware
//
#include <hip/hip_runtime.h>
#include <hip/hip_bf16.h>
#include <math.h>

typedef unsigned short u16;
typedef __attribute__((ext_vector_type(16))) __bf16 bf16x16;
typedef __attribute__((ext_vector_type(8)))  float  f32x8;

#define N_TOK 32768
#define DDIM  1024
#define HDIM  512
#define NEXP  8
#define MTILE 32          // tokens per block (two 16-row WMMA tiles, B-frag reuse x2)

__device__ __forceinline__ u16 f32_to_bf16(float f) {
  unsigned int u = __float_as_uint(f);
  u += 0x7FFFu + ((u >> 16) & 1u);          // round-to-nearest-even
  return (u16)(u >> 16);
}

union Frag32B { uint4 q[2]; bf16x16 v; };

// A fragment (16x32 bf16, row-major source, row stride = ldrow elements).
// lane<16: row=lane,  K = k0+0..7  and k0+16..23
// lane>=16: row=lane-16, K = k0+8..15 and k0+24..31
__device__ __forceinline__ bf16x16 load_a16(const u16* base, int lane, int k0, int ldrow) {
  int row = lane & 15;
  int kb  = k0 + ((lane >> 4) << 3);
  Frag32B f;
  const u16* p = base + row * ldrow + kb;
  f.q[0] = *(const uint4*)(p);
  f.q[1] = *(const uint4*)(p + 16);
  return f.v;
}

// B fragment (32x16 bf16) from N-major (pre-transposed) weight [Ntot x Ktot].
// lane<16: col=n0+lane, K=k0+0..7 and k0+16..23 ; lane>=16: col=n0+lane-16, K=k0+8..15 and k0+24..31
__device__ __forceinline__ bf16x16 load_b16(const u16* w, int lane, int n0, int k0, int ktot) {
  int n  = n0 + (lane & 15);
  int kb = k0 + ((lane >> 4) << 3);
  Frag32B f;
  const u16* p = w + (size_t)n * ktot + kb;
  f.q[0] = *(const uint4*)(p);
  f.q[1] = *(const uint4*)(p + 16);
  return f.v;
}

// CDNA5 async memory->LDS copy (16B per lane), tracked by ASYNCcnt.
__device__ __forceinline__ void async_copy_b128(unsigned lds_byte_off, const void* gaddr) {
  asm volatile("global_load_async_to_lds_b128 %0, %1, off"
               :: "v"(lds_byte_off), "v"(gaddr)
               : "memory");
}
__device__ __forceinline__ void wait_async0() {
  asm volatile("s_wait_asynccnt 0x0" ::: "memory");
}

__global__ void zero_counts_kernel(int* counts) {
  if (threadIdx.x < NEXP) counts[threadIdx.x] = 0;
}

// x f32 -> bf16, 8 elements per thread
__global__ void cvt_x_kernel(const float* __restrict__ x, u16* __restrict__ xb, int n8) {
  int i = blockIdx.x * blockDim.x + threadIdx.x;
  if (i >= n8) return;
  const float4* s = (const float4*)x;
  float4 a = s[2 * i], b = s[2 * i + 1];
  union { u16 u[8]; uint4 q; } r;
  r.u[0] = f32_to_bf16(a.x); r.u[1] = f32_to_bf16(a.y);
  r.u[2] = f32_to_bf16(a.z); r.u[3] = f32_to_bf16(a.w);
  r.u[4] = f32_to_bf16(b.x); r.u[5] = f32_to_bf16(b.y);
  r.u[6] = f32_to_bf16(b.z); r.u[7] = f32_to_bf16(b.w);
  ((uint4*)xb)[i] = r.q;
}

// src f32 [R x C] (batched via blockIdx.y) -> dst bf16 [C x R] (N-major for WMMA B)
__global__ void cvt_tr_kernel(const float* __restrict__ src, u16* __restrict__ dst, int R, int C) {
  size_t base = (size_t)blockIdx.y * (size_t)R * (size_t)C;
  int idx = blockIdx.x * blockDim.x + threadIdx.x;
  if (idx >= R * C) return;
  int r = idx / C, c = idx - r * C;
  dst[base + (size_t)c * R + r] = f32_to_bf16(src[base + idx]);
}

// One wave per token: logits = x@Wg + bias, top-2, softmax, append to expert lists
__global__ __launch_bounds__(256) void router_kernel(
    const float* __restrict__ x, const float* __restrict__ Wg, const float* __restrict__ gb,
    int* __restrict__ counts, int* __restrict__ lists, float* __restrict__ wlists) {
  int wave = threadIdx.x >> 5, lane = threadIdx.x & 31;
  int tok = blockIdx.x * 8 + wave;
  const float* xr = x + (size_t)tok * DDIM;
  float acc[NEXP];
#pragma unroll
  for (int e = 0; e < NEXP; ++e) acc[e] = 0.0f;
  for (int j = 0; j < DDIM / 32; ++j) {
    int d = lane + 32 * j;
    float xv = xr[d];
    const float4* wr = (const float4*)(Wg + (size_t)d * NEXP);
    float4 w0 = wr[0], w1 = wr[1];
    acc[0] += xv * w0.x; acc[1] += xv * w0.y; acc[2] += xv * w0.z; acc[3] += xv * w0.w;
    acc[4] += xv * w1.x; acc[5] += xv * w1.y; acc[6] += xv * w1.z; acc[7] += xv * w1.w;
  }
#pragma unroll
  for (int off = 16; off > 0; off >>= 1) {
#pragma unroll
    for (int e = 0; e < NEXP; ++e) acc[e] += __shfl_xor(acc[e], off, 32);
  }
  if (lane == 0) {
    float best = -3.4e38f, sec = -3.4e38f;
    int bi = 0, si = 0;
#pragma unroll
    for (int e = 0; e < NEXP; ++e) {
      float v = acc[e] + gb[e];
      if (v > best)      { sec = best; si = bi; best = v; bi = e; }
      else if (v > sec)  { sec = v; si = e; }
    }
    float e1 = expf(sec - best);
    float s  = 1.0f + e1;
    float w0 = 1.0f / s, w1 = e1 / s;
    int p0 = atomicAdd(&counts[bi], 1);
    lists[bi * N_TOK + p0] = tok; wlists[bi * N_TOK + p0] = w0;
    int p1 = atomicAdd(&counts[si], 1);
    lists[si * N_TOK + p1] = tok; wlists[si * N_TOK + p1] = w1;
  }
}

// Fused expert: out[32 tok x 1024] (+)= gate * (gelu(X@W1) @ W2 + b2)
// 8 waves; stage1: wave owns 64 H-cols x 32 rows; stage2: wave owns 128 out-cols x 32 rows.
// Each B fragment feeds two WMMAs (row tiles 0-15 / 16-31) -> half the global B traffic per FLOP.
template <bool ROUTED>
__global__ __launch_bounds__(256) void expert_kernel(
    const u16* __restrict__ xbf, const u16* __restrict__ w1t, const u16* __restrict__ w2t,
    const float* __restrict__ b2, const int* __restrict__ counts,
    const int* __restrict__ lists, const float* __restrict__ wlists,
    float* __restrict__ out) {
  __shared__ __align__(16) u16 lsX[MTILE * DDIM];   // 64KB gathered X tile (bf16)
  __shared__ __align__(16) u16 lsH[MTILE * HDIM];   // 32KB gelu(H) tile (bf16)
  __shared__ int   lsTok[MTILE];
  __shared__ float lsGate[MTILE];

  const int e = ROUTED ? (int)blockIdx.y : 0;
  const int tile = blockIdx.x;
  int cnt = N_TOK;
  if (ROUTED) {
    cnt = counts[e];
    if (tile * MTILE >= cnt) return;
  }
  const int t = threadIdx.x;
  if (t < MTILE) {
    int idx = tile * MTILE + t;
    int tok; float g;
    if (ROUTED) {
      if (idx < cnt) { tok = lists[e * N_TOK + idx]; g = wlists[e * N_TOK + idx]; }
      else           { tok = lists[e * N_TOK];       g = 0.0f; } // pad row, gate 0
    } else { tok = idx; g = 1.0f; }
    lsTok[t] = tok; lsGate[t] = g;
  }
  __syncthreads();

  { // async gather of 32 token rows (2KB each) into LDS: 256B per thread = 16 x b128
    int r = t >> 3, sub = t & 7;
    const char* src = (const char*)(xbf + (size_t)lsTok[r] * DDIM);
    unsigned dst = (unsigned)(size_t)(const void*)(lsX + r * DDIM);
#pragma unroll
    for (int j = 0; j < 16; ++j) {
      int boff = (sub + 8 * j) * 16;
      async_copy_b128(dst + boff, src + boff);
    }
  }
  wait_async0();
  __syncthreads();

  const int wave = t >> 5, lane = t & 31;
  const u16* W1 = w1t + (ROUTED ? (size_t)e * HDIM * DDIM : 0);
  const u16* W2 = w2t + (ROUTED ? (size_t)e * DDIM * HDIM : 0);
  const float* B2 = b2 + (ROUTED ? (size_t)e * DDIM : 0);

  // ---- Stage 1: H = gelu(X @ W1), K = 1024, this wave owns cols [wave*64, wave*64+64)
  {
    f32x8 acc[8] = {};   // [rowtile][j]: rt*4 + j
    for (int kk = 0; kk < DDIM; kk += 32) {
      bf16x16 a0 = load_a16(lsX,              lane, kk, DDIM);
      bf16x16 a1 = load_a16(lsX + 16 * DDIM,  lane, kk, DDIM);
#pragma unroll
      for (int j = 0; j < 4; ++j) {
        bf16x16 b = load_b16(W1, lane, wave * 64 + j * 16, kk, DDIM);
        acc[j]     = __builtin_amdgcn_wmma_f32_16x16x32_bf16(false, a0, false, b, (short)0, acc[j],     false, false);
        acc[4 + j] = __builtin_amdgcn_wmma_f32_16x16x32_bf16(false, a1, false, b, (short)0, acc[4 + j], false, false);
      }
    }
#pragma unroll
    for (int rt = 0; rt < 2; ++rt) {
#pragma unroll
      for (int j = 0; j < 4; ++j) {
        int col = wave * 64 + j * 16 + (lane & 15);
#pragma unroll
        for (int rr = 0; rr < 8; ++rr) {
          int row = rt * 16 + rr + ((lane >> 4) << 3);
          float v = acc[rt * 4 + j][rr];
          float gel = 0.5f * v * (1.0f + erff(v * 0.70710678118654752f)); // exact GELU
          lsH[row * HDIM + col] = f32_to_bf16(gel);
        }
      }
    }
  }
  __syncthreads();

  // ---- Stage 2: OUT = H @ W2 + b2, K = 512, this wave owns cols [wave*128, wave*128+128)
  f32x8 o[16] = {};      // [rowtile][j]: rt*8 + j
  for (int kk = 0; kk < HDIM; kk += 32) {
    bf16x16 a0 = load_a16(lsH,             lane, kk, HDIM);
    bf16x16 a1 = load_a16(lsH + 16 * HDIM, lane, kk, HDIM);
#pragma unroll
    for (int j = 0; j < 8; ++j) {
      bf16x16 b = load_b16(W2, lane, wave * 128 + j * 16, kk, HDIM);
      o[j]     = __builtin_amdgcn_wmma_f32_16x16x32_bf16(false, a0, false, b, (short)0, o[j],     false, false);
      o[8 + j] = __builtin_amdgcn_wmma_f32_16x16x32_bf16(false, a1, false, b, (short)0, o[8 + j], false, false);
    }
  }
#pragma unroll
  for (int rt = 0; rt < 2; ++rt) {
#pragma unroll
    for (int j = 0; j < 8; ++j) {
      int col = wave * 128 + j * 16 + (lane & 15);
      float bias = B2[col];
#pragma unroll
      for (int rr = 0; rr < 8; ++rr) {
        int row = rt * 16 + rr + ((lane >> 4) << 3);
        size_t oidx = (size_t)lsTok[row] * DDIM + col;
        float val = o[rt * 8 + j][rr] + bias;
        if (ROUTED) atomicAdd(out + oidx, lsGate[row] * val);
        else        out[oidx] = val;   // shared pass initializes d_out
      }
    }
  }
}

extern "C" void kernel_launch(void* const* d_in, const int* in_sizes, int n_in,
                              void* d_out, int out_size, void* d_ws, size_t ws_size,
                              hipStream_t stream) {
  const float* x   = (const float*)d_in[0];
  const float* Wg  = (const float*)d_in[1];
  const float* gb  = (const float*)d_in[2];
  const float* W1s = (const float*)d_in[3];
  const float* W2s = (const float*)d_in[4];
  const float* b2s = (const float*)d_in[5];
  const float* W1r = (const float*)d_in[6];
  const float* W2r = (const float*)d_in[7];
  const float* b2r = (const float*)d_in[8];
  float* out = (float*)d_out;
  (void)in_sizes; (void)n_in; (void)out_size; (void)ws_size;

  char* ws = (char*)d_ws;
  size_t off = 0;
  auto take = [&](size_t bytes) -> char* {
    char* p = ws + off;
    off = (off + bytes + 255) & ~(size_t)255;
    return p;
  };
  u16*   xbf    = (u16*)take((size_t)N_TOK * DDIM * 2);          // 64 MB
  u16*   w1st   = (u16*)take((size_t)HDIM * DDIM * 2);
  u16*   w2st   = (u16*)take((size_t)DDIM * HDIM * 2);
  u16*   w1rt   = (u16*)take((size_t)NEXP * HDIM * DDIM * 2);
  u16*   w2rt   = (u16*)take((size_t)NEXP * DDIM * HDIM * 2);
  int*   counts = (int*)take(NEXP * sizeof(int));
  int*   lists  = (int*)take((size_t)NEXP * N_TOK * sizeof(int));
  float* wl     = (float*)take((size_t)NEXP * N_TOK * sizeof(float));

  zero_counts_kernel<<<1, 32, 0, stream>>>(counts);

  int n8 = N_TOK * DDIM / 8;
  cvt_x_kernel<<<(n8 + 255) / 256, 256, 0, stream>>>(x, xbf, n8);

  int wgrid = (DDIM * HDIM + 255) / 256;
  cvt_tr_kernel<<<dim3(wgrid, 1),    256, 0, stream>>>(W1s, w1st, DDIM, HDIM);
  cvt_tr_kernel<<<dim3(wgrid, 1),    256, 0, stream>>>(W2s, w2st, HDIM, DDIM);
  cvt_tr_kernel<<<dim3(wgrid, NEXP), 256, 0, stream>>>(W1r, w1rt, DDIM, HDIM);
  cvt_tr_kernel<<<dim3(wgrid, NEXP), 256, 0, stream>>>(W2r, w2rt, HDIM, DDIM);

  router_kernel<<<N_TOK / 8, 256, 0, stream>>>(x, Wg, gb, counts, lists, wl);

  // shared expert: plain stores (initializes out), then routed: atomic accumulation
  expert_kernel<false><<<dim3(N_TOK / MTILE, 1),    256, 0, stream>>>(xbf, w1st, w2st, b2s, counts, lists, wl, out);
  expert_kernel<true ><<<dim3(N_TOK / MTILE, NEXP), 256, 0, stream>>>(xbf, w1rt, w2rt, b2r, counts, lists, wl, out);
}
